// MultiHeadedAttention_90383291777255
// MI455X (gfx1250) — compile-verified, matrix operand order checked by reading, not run on hardware
//
#include <hip/hip_runtime.h>

#define B_  4
#define S_  2048
#define D_  1024
#define H_  16
#define HD_ 64

typedef __attribute__((ext_vector_type(16))) __bf16        v16bf;
typedef __attribute__((ext_vector_type(8)))  float         v8f;
typedef __attribute__((ext_vector_type(4)))  int           v4i;
typedef __attribute__((ext_vector_type(8)))  int           v8i;
typedef __attribute__((ext_vector_type(4)))  unsigned int  v4u;

union FragA { unsigned int u[8]; v16bf v; };

#define LAS3 __attribute__((address_space(3)))

// hardware bf16 convert (v_cvt_bf16_f32 path)
static __device__ __forceinline__ unsigned short f2bf(float f) {
    union { __bf16 b; unsigned short s; } u;
    u.b = (__bf16)f;
    return u.s;
}
static __device__ __forceinline__ unsigned int pack2(float lo, float hi) {
    return (unsigned int)f2bf(lo) | ((unsigned int)f2bf(hi) << 16);
}
// k-index of element pair p (0..7) for the 16-bit 16x32 A / 32x16 B layouts
static __device__ __forceinline__ int kpair(int p, int half) {
    return ((p & 4) << 2) + half * 8 + ((p & 3) << 1);
}

// TDM: DMA a 32x64 bf16 tile (row stride 64 elems) into LDS with +16B row pad
// -> produces the [32][72]-ushort padded layout directly (pad after every
//    32 DWORDs (code 4), pad amount 4 DWORDs (code 3)).
static __device__ __forceinline__ void tdm_load_tile32x64(
        const unsigned short* gsrc, unsigned lds_addr) {
    unsigned long long ga = (unsigned long long)gsrc;
    v4u g0 = { 1u,                                   // count=1, user mode
               lds_addr,                             // D#.lds_addr (bytes)
               (unsigned)ga,                         // global_addr[31:0]
               (unsigned)((ga >> 32) & 0x01FFFFFFull) | 0x80000000u }; // [56:32] | type=2
    v8i g1 = { (int)((1u << 16) | (1u << 20) | (4u << 22) | (3u << 25)),
               // data_size=2B | pad_enable | pad_interval=32dw | pad_amount=4dw
               (int)(64u << 16),                     // tensor_dim0 = 64
               (int)(2048u << 16),                   // tensor_dim1 = 2048
               (int)(64u << 16),                     // tile_dim0 = 64
               32,                                   // tile_dim1 = 32
               64,                                   // tensor_dim0_stride = 64
               0, 0 };
    v4i gz4 = { 0, 0, 0, 0 };                        // 2-D tensor: groups 2/3 unused
    v8i gz8 = { 0, 0, 0, 0, 0, 0, 0, 0 };
    __builtin_amdgcn_tensor_load_to_lds(g0, g1, gz4, gz4, gz8, 0);
}

// ---------------------------------------------------------------------------
// Y[b,h,s,hd] (bf16) = (X[b,s,:] @ W + bias), head-split layout
// Block: 256 threads (8 waves). Block tile 128x128; wave tile 32x64. K-step 32.
// ---------------------------------------------------------------------------
__global__ __launch_bounds__(256)
void qkv_proj_kernel(const float* __restrict__ X, const float* __restrict__ W,
                     const float* __restrict__ bias, unsigned short* __restrict__ Y) {
    __shared__ unsigned short xs[128][34];   // X tile, bf16, [row][k]
    __shared__ unsigned short wt[128][34];   // W tile transposed, bf16, [n][k]

    const int tid  = threadIdx.x;
    const int lane = tid & 31, wave = tid >> 5;
    const int half = lane >> 4, l16 = lane & 15;
    const int mw = (wave & 3) * 32;          // wave row offset in block tile
    const int nw = (wave >> 2) * 64;         // wave col offset in block tile
    const int m0 = blockIdx.y * 128;
    const int n0 = blockIdx.x * 128;

    v8f acc[2][4] = {};

    for (int kt = 0; kt < D_ / 32; ++kt) {
        const int kk0 = kt * 32;
        __syncthreads();
        // X tile 128x32 f32 -> bf16 LDS (1024 float4, 4/thread)
        #pragma unroll
        for (int j = 0; j < 4; ++j) {
            int idx = tid + j * 256;
            int row = idx >> 3, seg = (idx & 7) * 4;
            const float4 xv = *(const float4*)(X + (size_t)(m0 + row) * D_ + kk0 + seg);
            *(unsigned int*)&xs[row][seg]     = pack2(xv.x, xv.y);
            *(unsigned int*)&xs[row][seg + 2] = pack2(xv.z, xv.w);
        }
        // W tile 32x128 f32 -> transposed bf16 LDS
        #pragma unroll
        for (int j = 0; j < 4; ++j) {
            int idx = tid + j * 256;
            int k = idx >> 5, nseg = (idx & 31) * 4;
            const float4 wv = *(const float4*)(W + (size_t)(kk0 + k) * D_ + n0 + nseg);
            wt[nseg + 0][k] = f2bf(wv.x);
            wt[nseg + 1][k] = f2bf(wv.y);
            wt[nseg + 2][k] = f2bf(wv.z);
            wt[nseg + 3][k] = f2bf(wv.w);
        }
        __syncthreads();

        FragA a[2];
        #pragma unroll
        for (int rt = 0; rt < 2; ++rt)
            #pragma unroll
            for (int p = 0; p < 8; ++p)
                a[rt].u[p] = *(const unsigned int*)&xs[mw + rt * 16 + l16][kpair(p, half)];
        #pragma unroll
        for (int jn = 0; jn < 4; ++jn) {
            FragA b;
            #pragma unroll
            for (int p = 0; p < 8; ++p)
                b.u[p] = *(const unsigned int*)&wt[nw + jn * 16 + l16][kpair(p, half)];
            #pragma unroll
            for (int rt = 0; rt < 2; ++rt)
                acc[rt][jn] = __builtin_amdgcn_wmma_f32_16x16x32_bf16(
                    false, a[rt].v, false, b.v, (short)0, acc[rt][jn], false, false);
        }
    }

    // epilogue: +bias, bf16, head-split store
    #pragma unroll
    for (int jn = 0; jn < 4; ++jn) {
        int ng = n0 + nw + jn * 16 + l16;
        float bv = bias[ng];
        int h = ng >> 6, hd = ng & (HD_ - 1);
        #pragma unroll
        for (int rt = 0; rt < 2; ++rt)
            #pragma unroll
            for (int r = 0; r < 8; ++r) {
                int mg = m0 + mw + rt * 16 + r + half * 8;
                int bb = mg >> 11, s = mg & (S_ - 1);
                Y[(((size_t)(bb * H_ + h)) * S_ + s) * HD_ + hd] = f2bf(acc[rt][jn][r] + bv);
            }
    }
}

// ---------------------------------------------------------------------------
// Flash attention: 256 threads = 8 waves, 128 q-rows per (b,h) block.
// K and V chunks (32 keys) DMA'd into LDS by the Tensor Data Mover.
// ---------------------------------------------------------------------------
__global__ __launch_bounds__(256)
void attn_kernel(const unsigned short* __restrict__ Q,
                 const unsigned short* __restrict__ K,
                 const unsigned short* __restrict__ V,
                 const unsigned char* __restrict__ mask,
                 float* __restrict__ out) {
    __shared__ unsigned short ks[32][72];       // K chunk [key][d], bf16 (TDM-padded)
    __shared__ unsigned short vr[32][72];       // V chunk [key][d], bf16 (TDM-padded)
    __shared__ unsigned short ps[8 * 16 * 32];  // per-wave P tiles (16x32)

    const int tid = threadIdx.x, lane = tid & 31, wave = tid >> 5;
    const int half = lane >> 4, l16 = lane & 15;
    const int bh = blockIdx.y;                  // b*H + h
    const int bb = bh >> 4, h = bh & (H_ - 1);
    const int q0 = blockIdx.x * 128 + wave * 16;
    const size_t base = (size_t)bh * S_ * HD_;
    const unsigned short* qp = Q + base;
    const unsigned short* kp = K + base;
    const unsigned short* vp = V + base;

    const unsigned ks_lds = (unsigned)(unsigned long long)(LAS3 unsigned short*)&ks[0][0];
    const unsigned vr_lds = (unsigned)(unsigned long long)(LAS3 unsigned short*)&vr[0][0];

    // preload Q A-fragments: 16x64 = 2 chunks of 16x32
    FragA qa[2];
    #pragma unroll
    for (int c = 0; c < 2; ++c)
        #pragma unroll
        for (int p = 0; p < 8; ++p)
            qa[c].u[p] = *(const unsigned int*)(qp + (size_t)(q0 + l16) * HD_ +
                                                c * 32 + kpair(p, half));

    v8f acc[4] = {};
    float mrow[8], lrow[8];
    #pragma unroll
    for (int r = 0; r < 8; ++r) { mrow[r] = -1e30f; lrow[r] = 0.f; }

    for (int t = 0; t < S_ / 32; ++t) {
        const int kb = t * 32;
        __syncthreads();
        // TDM: one wave issues both tile DMAs; EXEC-independent, SGPR descriptors
        if (wave == 0) {
            tdm_load_tile32x64(kp + (size_t)kb * HD_, ks_lds);
            tdm_load_tile32x64(vp + (size_t)kb * HD_, vr_lds);
        }
        if (t + 1 < S_ / 32) {   // prefetch next chunk -> global_prefetch_b8
            __builtin_prefetch(kp + (size_t)(kb + 32) * HD_, 0, 1);
            __builtin_prefetch(vp + (size_t)(kb + 32) * HD_, 0, 1);
        }
        if (wave == 0) __builtin_amdgcn_s_wait_tensorcnt(0);
        __syncthreads();

        // scores: two 16x16 tiles over 32 keys
        v8f sc[2];
        #pragma unroll
        for (int nh = 0; nh < 2; ++nh) {
            v8f c = {};
            #pragma unroll
            for (int cc = 0; cc < 2; ++cc) {
                FragA kf;
                #pragma unroll
                for (int p = 0; p < 8; ++p)
                    kf.u[p] = *(const unsigned int*)&ks[nh * 16 + l16][cc * 32 + kpair(p, half)];
                c = __builtin_amdgcn_wmma_f32_16x16x32_bf16(
                    false, qa[cc].v, false, kf.v, (short)0, c, false, false);
            }
            sc[nh] = c;
        }

        // scale + mask (mask True => masked out)
        const float sscale = 0.125f;            // 1/sqrt(64)
        int mk0 = mask[(size_t)bb * S_ + kb + l16];
        int mk1 = mask[(size_t)bb * S_ + kb + 16 + l16];
        #pragma unroll
        for (int r = 0; r < 8; ++r) {
            sc[0][r] = mk0 ? -1e30f : sc[0][r] * sscale;
            sc[1][r] = mk1 ? -1e30f : sc[1][r] * sscale;
        }

        // online softmax (rows live across 16-lane halves; reduce via shfl_xor)
        float alpha[8];
        #pragma unroll
        for (int r = 0; r < 8; ++r) {
            float x = fmaxf(sc[0][r], sc[1][r]);
            x = fmaxf(x, __shfl_xor(x, 1));
            x = fmaxf(x, __shfl_xor(x, 2));
            x = fmaxf(x, __shfl_xor(x, 4));
            x = fmaxf(x, __shfl_xor(x, 8));
            float mn = fmaxf(mrow[r], x);
            float a  = __expf(mrow[r] - mn);
            float p0 = __expf(sc[0][r] - mn);
            float p1 = __expf(sc[1][r] - mn);
            float sm = p0 + p1;
            sm += __shfl_xor(sm, 1); sm += __shfl_xor(sm, 2);
            sm += __shfl_xor(sm, 4); sm += __shfl_xor(sm, 8);
            lrow[r] = lrow[r] * a + sm;
            mrow[r] = mn;
            alpha[r] = a;
            int prow = wave * 16 + r + half * 8;
            ps[prow * 32 + l16]      = f2bf(p0);
            ps[prow * 32 + 16 + l16] = f2bf(p1);
        }
        // LDS ops from the same wave are in-order (ISA 7.3); compiler barrier only
        asm volatile("" ::: "memory");

        FragA pa;
        #pragma unroll
        for (int p = 0; p < 8; ++p)
            pa.u[p] = *(const unsigned int*)&ps[(wave * 16 + l16) * 32 + kpair(p, half)];

        // acc = acc*alpha + P @ V  (V gathered transposed from row-major LDS)
        #pragma unroll
        for (int jd = 0; jd < 4; ++jd) {
            #pragma unroll
            for (int r = 0; r < 8; ++r) acc[jd][r] *= alpha[r];
            FragA vb;
            const int d = jd * 16 + l16;
            #pragma unroll
            for (int p = 0; p < 8; ++p) {
                int k0 = kpair(p, half);
                unsigned lo = vr[k0][d], hi = vr[k0 + 1][d];
                vb.u[p] = lo | (hi << 16);
            }
            acc[jd] = __builtin_amdgcn_wmma_f32_16x16x32_bf16(
                false, pa.v, false, vb.v, (short)0, acc[jd], false, false);
        }
    }

    // out[b, s, h*64 + d] = acc / l
    #pragma unroll
    for (int r = 0; r < 8; ++r) {
        float inv = 1.0f / lrow[r];
        int s = q0 + r + half * 8;
        #pragma unroll
        for (int jd = 0; jd < 4; ++jd) {
            int d = jd * 16 + l16;
            out[((size_t)bb * S_ + s) * D_ + h * HD_ + d] = acc[jd][r] * inv;
        }
    }
}

extern "C" void kernel_launch(void* const* d_in, const int* in_sizes, int n_in,
                              void* d_out, int out_size, void* d_ws, size_t ws_size,
                              hipStream_t stream) {
    const float* query         = (const float*)d_in[0];
    const float* key           = (const float*)d_in[1];
    const float* value         = (const float*)d_in[2];
    const unsigned char* mask  = (const unsigned char*)d_in[3];
    const float* Wq            = (const float*)d_in[4];
    const float* bq            = (const float*)d_in[5];
    const float* Wk            = (const float*)d_in[6];
    const float* bk            = (const float*)d_in[7];
    const float* Wv            = (const float*)d_in[8];
    const float* bv            = (const float*)d_in[9];
    float* out                 = (float*)d_out;

    const size_t elems = (size_t)B_ * S_ * D_;
    unsigned short* wsq = (unsigned short*)d_ws;     // bf16 q [B,H,S,HD]
    unsigned short* wsk = wsq + elems;               // bf16 k
    unsigned short* wsv = wsk + elems;               // bf16 v

    dim3 pg(D_ / 128, (B_ * S_) / 128);              // (8, 64)
    qkv_proj_kernel<<<pg, 256, 0, stream>>>(query, Wq, bq, wsq);
    qkv_proj_kernel<<<pg, 256, 0, stream>>>(key,   Wk, bk, wsk);
    qkv_proj_kernel<<<pg, 256, 0, stream>>>(value, Wv, bv, wsv);

    dim3 ag(S_ / 128, B_ * H_);                      // (16, 64)
    attn_kernel<<<ag, 256, 0, stream>>>(wsq, wsk, wsv, mask, out);
}
